// LiquidNCPNetwork_50826642981150
// MI455X (gfx1250) — matrix operand order repaired and checked by reading, not compiled, not run on hardware
//
#include <hip/hip_runtime.h>
#include <hip/hip_bf16.h>

// ---------------- problem constants ----------------
#define IN_F    64
#define N_INTER 154
#define N_CMD   102
#define N_OUT   16
#define BATCHSZ 128
#define SEQLEN  1024

// padded K dims (multiples of 32 for 16x16x32 bf16 WMMA)
#define KPI 224   // inter head 218 -> 224 (7 k-tiles)
#define KPC 256   // command head 256      (8 k-tiles)
#define KPM 128   // motor head 118 -> 128 (4 k-tiles)
#define KTI 7
#define KTC 8
#define KTM 4
#define NTI 10    // ceil(154/16)
#define NTC 7     // ceil(102/16)

// fused weight matrix strides (elements), 3 matrices per layer: g, h, f(=fg+fh)
#define WI_MAT (N_INTER * KPI)   // 34496
#define WC_MAT (N_CMD   * KPC)   // 26112
#define WM_MAT (N_OUT   * KPM)   // 2048

// workspace byte offsets (bf16 weights then f32 fused biases)
#define WS_WI 0u
#define WS_WC 206976u            // 3*154*224*2
#define WS_WM 363648u            // + 3*102*256*2
#define WS_BI 375936u            // + 3*16*128*2
#define WS_BC 377784u            // + 3*154*4
#define WS_BM 379008u            // + 3*102*4   (total 379200 B)

// LDS carve offsets (bytes)
#define L_WI   0u
#define L_WM   206976u
#define L_INPI 219264u
#define L_INPC 226432u
#define L_INPM 234624u
#define L_BI   238720u
#define L_BC   240640u
#define L_BM   241984u
#define SMEM_BYTES 242176u

typedef __attribute__((ext_vector_type(16))) __bf16 v16bf;
typedef __attribute__((ext_vector_type(8)))  float  v8f;
typedef __attribute__((ext_vector_type(4)))  float  f32x4;   // POD, union-safe

union FragB { v16bf v; f32x4 f4[2]; };

__device__ __forceinline__ float dsigm(float x) { return __fdividef(1.f, 1.f + __expf(-x)); }
__device__ __forceinline__ float dtanh(float x) { return 2.f * dsigm(2.f * x) - 1.f; }

// ---------------- prep: fuse (Wfg+Wfh), apply mask, cast to bf16, pad K ----------------
__global__ void ncp_prep(const float* __restrict__ Wg,  const float* __restrict__ bg,
                         const float* __restrict__ Wh,  const float* __restrict__ bh,
                         const float* __restrict__ Wfg, const float* __restrict__ bfg,
                         const float* __restrict__ Wfh, const float* __restrict__ bfh,
                         const float* __restrict__ mask,
                         int n_hid, int head, int k_pad,
                         __hip_bfloat16* outW_, float* outB)
{
    __bf16* outW = (__bf16*)outW_;
    int idx = blockIdx.x * blockDim.x + threadIdx.x;
    int wtot = 3 * n_hid * k_pad;
    if (idx < wtot) {
        int j = idx / (n_hid * k_pad);
        int r = idx % (n_hid * k_pad);
        int n = r / k_pad, k = r % k_pad;
        float v = 0.f;
        if (k < head) {
            float m = mask[n * head + k];
            float w = (j == 0) ? Wg[n * head + k]
                    : (j == 1) ? Wh[n * head + k]
                               : (Wfg[n * head + k] + Wfh[n * head + k]);
            v = w * m;
        }
        outW[idx] = (__bf16)v;
    } else if (idx < wtot + 3 * n_hid) {
        int r = idx - wtot;
        int j = r / n_hid, n = r % n_hid;
        outB[r] = (j == 0) ? bg[n] : (j == 1) ? bh[n] : (bfg[n] + bfh[n]);
    }
}

// ---------------- persistent recurrent kernel: 8 WGs x 16 batch rows ----------------
__global__ __launch_bounds__(256) void ncp_main(
    const float* __restrict__ x, const float* __restrict__ hidden,
    const __hip_bfloat16* wWi_, const __hip_bfloat16* wWc_, const __hip_bfloat16* wWm_,
    const float* __restrict__ wBi, const float* __restrict__ wBc, const float* __restrict__ wBm,
    float* __restrict__ out)
{
    const __bf16* wWi = (const __bf16*)wWi_;
    const __bf16* wWc = (const __bf16*)wWc_;
    const __bf16* wWm = (const __bf16*)wWm_;

    extern __shared__ char smem[];
    __bf16* sWi   = (__bf16*)(smem + L_WI);     // inter weights (3 x 154 x 224)
    __bf16* sWm   = (__bf16*)(smem + L_WM);     // motor weights (3 x 16 x 128)
    __bf16* sInpI = (__bf16*)(smem + L_INPI);   // [16][224] = x | h_i | pad
    __bf16* sInpC = (__bf16*)(smem + L_INPC);   // [16][256] = h_i | h_c
    __bf16* sInpM = (__bf16*)(smem + L_INPM);   // [16][128] = h_c | h_m | pad
    float*  sBi   = (float*)(smem + L_BI);      // [3][160] padded biases
    float*  sBc   = (float*)(smem + L_BC);      // [3][112]
    float*  sBm   = (float*)(smem + L_BM);      // [3][16]

    const int tid  = threadIdx.x;
    const int wave = tid >> 5;
    const int lane = tid & 31;
    const int col  = lane & 15;     // N (or M for A-frags) index within tile
    const int half = lane >> 4;     // lane group 0/1
    const int wgB  = blockIdx.x * 16;

    // motor cell for timestep tt (runs on wave 0 only), then h_c(tt) recurrent copy
    auto motor_and_copy = [&](int tt) {
        v8f ag = {}, ah = {}, af = {};
        #pragma unroll
        for (int kt = 0; kt < KTM; ++kt) {
            FragB a, b0, b1, b2;
            const __bf16* ap = sInpM + col * KPM + kt * 32 + 16 * half;
            a.f4[0] = *(const f32x4*)ap; a.f4[1] = *(const f32x4*)(ap + 8);
            const __bf16* bp = sWm + col * KPM + kt * 32 + 16 * half;
            b0.f4[0] = *(const f32x4*)bp;                 b0.f4[1] = *(const f32x4*)(bp + 8);
            b1.f4[0] = *(const f32x4*)(bp + WM_MAT);      b1.f4[1] = *(const f32x4*)(bp + WM_MAT + 8);
            b2.f4[0] = *(const f32x4*)(bp + 2 * WM_MAT);  b2.f4[1] = *(const f32x4*)(bp + 2 * WM_MAT + 8);
            ag = __builtin_amdgcn_wmma_f32_16x16x32_bf16(false, a.v, false, b0.v, (short)0, ag, false, false);
            ah = __builtin_amdgcn_wmma_f32_16x16x32_bf16(false, a.v, false, b1.v, (short)0, ah, false, false);
            af = __builtin_amdgcn_wmma_f32_16x16x32_bf16(false, a.v, false, b2.v, (short)0, af, false, false);
        }
        float bg = sBm[col], bh = sBm[16 + col], bfv = sBm[32 + col];
        #pragma unroll
        for (int v = 0; v < 8; ++v) {
            int m = v + 8 * half;
            float g  = dtanh(ag[v] + bg);
            float hc = dtanh(ah[v] + bh);
            float s  = dsigm(af[v] + bfv);
            float o  = g * (1.f - s) + s * hc;
            out[((size_t)(wgB + m) * SEQLEN + tt) * N_OUT + col] = o;   // y_tt (f32)
            sInpM[m * KPM + N_CMD + col] = (__bf16)o;                   // h_m recurrent update
        }
        // h_c(tt) into command-input recurrent slot (consumed by cmd(tt+1) after barrier)
        for (int i = lane; i < 16 * N_CMD; i += 32) {
            int m = i / N_CMD, c = i % N_CMD;
            sInpC[m * KPC + N_INTER + c] = sInpM[m * KPM + c];
        }
    };

    // ---- one-time staging: weights/biases -> LDS, x_0, init recurrent state ----
    { const f32x4* s = (const f32x4*)wWi; f32x4* d = (f32x4*)sWi;
      for (int i = tid; i < 12936; i += 256) d[i] = s[i]; }
    { const f32x4* s = (const f32x4*)wWm; f32x4* d = (f32x4*)sWm;
      for (int i = tid; i < 768; i += 256) d[i] = s[i]; }
    for (int i = tid; i < 480; i += 256) { int j = i / 160, n = i % 160; sBi[i] = (n < N_INTER) ? wBi[j * N_INTER + n] : 0.f; }
    for (int i = tid; i < 336; i += 256) { int j = i / 112, n = i % 112; sBc[i] = (n < N_CMD)   ? wBc[j * N_CMD + n]   : 0.f; }
    for (int i = tid; i < 48;  i += 256) sBm[i] = wBm[i];
    for (int i = tid; i < 4864; i += 256) ((int*)sInpI)[i] = 0;   // zero all act buffers (incl. K padding)
    __syncthreads();
    for (int i = tid; i < 16 * 272; i += 256) {
        int m = i / 272, c = i % 272;
        __bf16 v = (__bf16)hidden[(size_t)(wgB + m) * 272 + c];
        if (c < N_INTER)                { sInpI[m * KPI + IN_F + c] = v; sInpC[m * KPC + c] = v; }
        else if (c < N_INTER + N_CMD)   { int cc = c - N_INTER; sInpC[m * KPC + N_INTER + cc] = v; sInpM[m * KPM + cc] = v; }
        else                            { sInpM[m * KPM + N_CMD + (c - N_INTER - N_CMD)] = v; }
    }
    for (int i = tid; i < 16 * IN_F; i += 256) {                  // stage x_0
        int m = i >> 6, f = i & 63;
        sInpI[m * KPI + f] = (__bf16)x[((size_t)(wgB + m) * SEQLEN) * IN_F + f];
    }

    // ---- command-layer weights: register-resident for the whole scan ----
    // Each wave (0..6) owns one 16-col N-tile; its 24 B-fragments (3 mats x 8 k-tiles,
    // 192 VGPRs) are invariant across time, so load them ONCE from L2 and never again.
    FragB cb[KTC][3];
    if (wave < 7) {
        int ng  = wave * 16 + col;
        int ncl = (ng < N_CMD) ? ng : (N_CMD - 1);
        const __bf16* bbase = wWc + (size_t)ncl * KPC + 16 * half;
        #pragma unroll
        for (int kt = 0; kt < KTC; ++kt) {
            #pragma unroll
            for (int j = 0; j < 3; ++j) {
                const __bf16* q = bbase + kt * 32 + j * WC_MAT;
                cb[kt][j].f4[0] = *(const f32x4*)q;
                cb[kt][j].f4[1] = *(const f32x4*)(q + 8);
            }
        }
    }
    __syncthreads();

    // ---- sequential scan: 2 phases / 2 barriers per step; motor pipelined 1 step back ----
    for (int t = 0; t < SEQLEN; ++t) {
        // phase 1: wave 0 = motor(t-1) + h_c copy (hidden under inter);
        //          waves 1..7 = inter(t), 10 N-tiles, weights in LDS
        if (wave == 0) {
            if (t > 0) motor_and_copy(t - 1);
        } else {
            for (int nt = wave - 1; nt < NTI; nt += 7) {
                v8f ag = {}, ah = {}, af = {};
                int ng  = nt * 16 + col;
                int ncl = (ng < N_INTER) ? ng : (N_INTER - 1);
                #pragma unroll
                for (int kt = 0; kt < KTI; ++kt) {
                    FragB a, b0, b1, b2;
                    const __bf16* ap = sInpI + col * KPI + kt * 32 + 16 * half;
                    a.f4[0] = *(const f32x4*)ap; a.f4[1] = *(const f32x4*)(ap + 8);
                    const __bf16* bp = sWi + ncl * KPI + kt * 32 + 16 * half;
                    b0.f4[0] = *(const f32x4*)bp;                 b0.f4[1] = *(const f32x4*)(bp + 8);
                    b1.f4[0] = *(const f32x4*)(bp + WI_MAT);      b1.f4[1] = *(const f32x4*)(bp + WI_MAT + 8);
                    b2.f4[0] = *(const f32x4*)(bp + 2 * WI_MAT);  b2.f4[1] = *(const f32x4*)(bp + 2 * WI_MAT + 8);
                    ag = __builtin_amdgcn_wmma_f32_16x16x32_bf16(false, a.v, false, b0.v, (short)0, ag, false, false);
                    ah = __builtin_amdgcn_wmma_f32_16x16x32_bf16(false, a.v, false, b1.v, (short)0, ah, false, false);
                    af = __builtin_amdgcn_wmma_f32_16x16x32_bf16(false, a.v, false, b2.v, (short)0, af, false, false);
                }
                float bg = sBi[ng], bh = sBi[160 + ng], bfv = sBi[320 + ng];
                bool valid = ng < N_INTER;
                #pragma unroll
                for (int v = 0; v < 8; ++v) {
                    int m = v + 8 * half;
                    float g  = dtanh(ag[v] + bg);
                    float hc = dtanh(ah[v] + bh);
                    float s  = dsigm(af[v] + bfv);
                    float o  = g * (1.f - s) + s * hc;
                    if (valid) sInpC[m * KPC + ng] = (__bf16)o;   // new h_i -> command input
                }
            }
        }
        __syncthreads();

        // phase 2: waves 0..6 = command(t) from register-resident weights;
        //          wave 7 = h_i recurrent copy + stage x_{t+1}
        if (wave < 7) {
            int ng = wave * 16 + col;
            v8f ag = {}, ah = {}, af = {};
            #pragma unroll
            for (int kt = 0; kt < KTC; ++kt) {
                FragB a;
                const __bf16* ap = sInpC + col * KPC + kt * 32 + 16 * half;
                a.f4[0] = *(const f32x4*)ap; a.f4[1] = *(const f32x4*)(ap + 8);
                ag = __builtin_amdgcn_wmma_f32_16x16x32_bf16(false, a.v, false, cb[kt][0].v, (short)0, ag, false, false);
                ah = __builtin_amdgcn_wmma_f32_16x16x32_bf16(false, a.v, false, cb[kt][1].v, (short)0, ah, false, false);
                af = __builtin_amdgcn_wmma_f32_16x16x32_bf16(false, a.v, false, cb[kt][2].v, (short)0, af, false, false);
            }
            float bg = sBc[ng], bh = sBc[112 + ng], bfv = sBc[224 + ng];
            bool valid = ng < N_CMD;
            #pragma unroll
            for (int v = 0; v < 8; ++v) {
                int m = v + 8 * half;
                float g  = dtanh(ag[v] + bg);
                float hc = dtanh(ah[v] + bh);
                float s  = dsigm(af[v] + bfv);
                float o  = g * (1.f - s) + s * hc;
                if (valid) sInpM[m * KPM + ng] = (__bf16)o;   // new h_c -> motor input
            }
        } else {
            for (int i = lane; i < 16 * N_INTER; i += 32) {
                int m = i / N_INTER, c = i % N_INTER;
                sInpI[m * KPI + IN_F + c] = sInpC[m * KPC + c];   // h_i recurrent update
            }
            if (t + 1 < SEQLEN) {
                for (int i = lane; i < 16 * IN_F; i += 32) {       // stage x_{t+1}
                    int m = i >> 6, f = i & 63;
                    const float* xp = x + ((size_t)(wgB + m) * SEQLEN + t + 1) * IN_F + f;
                    sInpI[m * KPI + f] = (__bf16)(*xp);
                    if (t + 2 < SEQLEN) __builtin_prefetch(xp + IN_F, 0, 1);
                }
            }
        }
        __syncthreads();
    }

    // trailing motor for the final timestep (y_{SEQ-1}, final h_m)
    if (wave == 0) motor_and_copy(SEQLEN - 1);
    __syncthreads();

    // ---- h_final = [h_i | h_c | h_m] (128 x 272), appended after ys ----
    // h_i from sInpC[0..153]; h_c from sInpM[0..101]; h_m from sInpM[102..117]
    const size_t YS = (size_t)BATCHSZ * SEQLEN * N_OUT;
    for (int i = tid; i < 16 * 272; i += 256) {
        int m = i / 272, c = i % 272;
        float v;
        if      (c < N_INTER)         v = (float)sInpC[m * KPC + c];
        else if (c < N_INTER + N_CMD) v = (float)sInpM[m * KPM + (c - N_INTER)];
        else                          v = (float)sInpM[m * KPM + N_CMD + (c - N_INTER - N_CMD)];
        out[YS + (size_t)(wgB + m) * 272 + c] = v;
    }
}

// ---------------- host launcher ----------------
extern "C" void kernel_launch(void* const* d_in, const int* in_sizes, int n_in,
                              void* d_out, int out_size, void* d_ws, size_t ws_size,
                              hipStream_t stream)
{
    (void)in_sizes; (void)n_in; (void)out_size; (void)ws_size;
    const float* x      = (const float*)d_in[0];
    const float* hidden = (const float*)d_in[1];

    char* ws = (char*)d_ws;
    __hip_bfloat16* wWi = (__hip_bfloat16*)(ws + WS_WI);
    __hip_bfloat16* wWc = (__hip_bfloat16*)(ws + WS_WC);
    __hip_bfloat16* wWm = (__hip_bfloat16*)(ws + WS_WM);
    float* wBi = (float*)(ws + WS_BI);
    float* wBc = (float*)(ws + WS_BC);
    float* wBm = (float*)(ws + WS_BM);

    // layer param blocks: [Wg,bg, Wh,bh, Wfg,bfg, Wfh,bfh, mask]
    const float* const* P = (const float* const*)d_in;

    int tot;
    tot = 3 * N_INTER * KPI + 3 * N_INTER;
    ncp_prep<<<(tot + 255) / 256, 256, 0, stream>>>(
        P[2], P[3], P[4], P[5], P[6], P[7], P[8], P[9], P[10],
        N_INTER, IN_F + N_INTER, KPI, wWi, wBi);

    tot = 3 * N_CMD * KPC + 3 * N_CMD;
    ncp_prep<<<(tot + 255) / 256, 256, 0, stream>>>(
        P[11], P[12], P[13], P[14], P[15], P[16], P[17], P[18], P[19],
        N_CMD, N_INTER + N_CMD, KPC, wWc, wBc);

    tot = 3 * N_OUT * KPM + 3 * N_OUT;
    ncp_prep<<<(tot + 255) / 256, 256, 0, stream>>>(
        P[20], P[21], P[22], P[23], P[24], P[25], P[26], P[27], P[28],
        N_OUT, N_CMD + N_OUT, KPM, wWm, wBm);

    ncp_main<<<dim3(BATCHSZ / 16), dim3(256), SMEM_BYTES, stream>>>(
        x, hidden, wWi, wWc, wWm, wBi, wBc, wBm, (float*)d_out);
}
